// SparseBottleneck_48249662603782
// MI455X (gfx1250) — matrix operand order repaired and checked by reading, not compiled, not run on hardware
//
#include <hip/hip_runtime.h>
#include <hip/hip_bf16.h>

// Problem constants (match reference setup_inputs)
#define B_    8
#define C_    256
#define H_    1024
#define T_    8192
#define KSP   512
#define NBINS 4096
#define BINLO (-16.0f)
#define BINW  (1.0f/128.0f)
#define PITCH 48   // LDS pitch in halves: 96B rows, keeps 16B alignment for b128 reads

typedef __attribute__((ext_vector_type(16))) __bf16 v16bf;
typedef __attribute__((ext_vector_type(8)))  float  v8f;

__device__ __forceinline__ unsigned short f2bf(float f) {
  unsigned u = __float_as_uint(f);
  u += 0x7FFFu + ((u >> 16) & 1u);   // round-to-nearest-even
  return (unsigned short)(u >> 16);
}
__device__ __forceinline__ float bf2f(unsigned short h) {
  return __uint_as_float(((unsigned)h) << 16);
}
__device__ __forceinline__ int binOf(float v) {
  int bin = (int)floorf((v - BINLO) * (1.0f / BINW));
  return bin < 0 ? 0 : (bin > NBINS - 1 ? NBINS - 1 : bin);
}

// ---------------- K0: init workspace ----------------
__global__ void init_ws(unsigned* __restrict__ hist, float* __restrict__ sume,
                        int* __restrict__ tiecnt) {
  int i = blockIdx.x * 256 + threadIdx.x;
  if (i < B_ * NBINS) hist[i] = 0u;
  if (i < B_) { sume[i] = 0.0f; tiecnt[i] = 0; }
}

// ---------------- K1: saliency GEMM (bf16 hi/lo WMMA) + histogram ----------------
__global__ __launch_bounds__(256)
void sal_gemm_hist(const float* __restrict__ x, const float* __restrict__ wsal,
                   const float* __restrict__ bsal, float* __restrict__ sal,
                   unsigned* __restrict__ ghist) {
  __shared__ unsigned short sAhi[128 * PITCH], sAlo[128 * PITCH];
  __shared__ unsigned short sBhi[64 * PITCH],  sBlo[64 * PITCH];
  __shared__ unsigned sHist[NBINS];

  const int tid = threadIdx.x;
  const int b  = blockIdx.z;
  const int m0 = blockIdx.y * 128;   // H tile
  const int n0 = blockIdx.x * 64;    // T tile

  for (int i = tid; i < NBINS; i += 256) sHist[i] = 0u;

  const int wid = tid >> 5, lane = tid & 31;
  const int wm = wid & 3, wn = wid >> 2;         // 4x2 wave grid, 32x32 per wave
  const int l = lane & 15, h2 = lane >> 4;

  v8f acc[2][2] = {};
  const float* xb = x + (size_t)b * C_ * T_;

  for (int kk = 0; kk < C_; kk += 32) {
    __syncthreads();
    // Stage A: w_sal tile [128 x 32] fp32 -> bf16 hi/lo, row-major in LDS
    #pragma unroll
    for (int p = 0; p < 4; ++p) {
      int m  = (tid >> 3) + p * 32;
      int kc = (tid & 7) * 4;
      float4 f = *(const float4*)(wsal + (size_t)(m0 + m) * C_ + kk + kc);
      unsigned short a0 = f2bf(f.x), a1 = f2bf(f.y), a2 = f2bf(f.z), a3 = f2bf(f.w);
      unsigned short c0 = f2bf(f.x - bf2f(a0)), c1 = f2bf(f.y - bf2f(a1));
      unsigned short c2 = f2bf(f.z - bf2f(a2)), c3 = f2bf(f.w - bf2f(a3));
      unsigned* dh = (unsigned*)&sAhi[m * PITCH + kc];
      unsigned* dl = (unsigned*)&sAlo[m * PITCH + kc];
      dh[0] = (unsigned)a0 | ((unsigned)a1 << 16);
      dh[1] = (unsigned)a2 | ((unsigned)a3 << 16);
      dl[0] = (unsigned)c0 | ((unsigned)c1 << 16);
      dl[1] = (unsigned)c2 | ((unsigned)c3 << 16);
    }
    // Stage B: x tile [32(K) x 64(N)] fp32 -> bf16 hi/lo, column(N)-major in LDS
    #pragma unroll
    for (int p = 0; p < 2; ++p) {
      int c  = (tid >> 4) + p * 16;
      int nt = (tid & 15) * 4;
      float4 f = *(const float4*)(xb + (size_t)(kk + c) * T_ + n0 + nt);
      float v[4] = {f.x, f.y, f.z, f.w};
      #pragma unroll
      for (int j = 0; j < 4; ++j) {
        unsigned short hh = f2bf(v[j]);
        sBhi[(nt + j) * PITCH + c] = hh;
        sBlo[(nt + j) * PITCH + c] = f2bf(v[j] - bf2f(hh));
      }
    }
    __syncthreads();

    union Frag { uint4 q[2]; v16bf v; };
    Frag ah[2], al[2], bh[2], bl[2];
    #pragma unroll
    for (int tm = 0; tm < 2; ++tm) {
      int row = wm * 32 + tm * 16 + l;
      int o1 = row * PITCH + h2 * 8;  // halves K {0..7 | 8..15}
      int o2 = o1 + 16;               // halves K {16..23 | 24..31}
      ah[tm].q[0] = *(const uint4*)&sAhi[o1];  ah[tm].q[1] = *(const uint4*)&sAhi[o2];
      al[tm].q[0] = *(const uint4*)&sAlo[o1];  al[tm].q[1] = *(const uint4*)&sAlo[o2];
    }
    #pragma unroll
    for (int tn = 0; tn < 2; ++tn) {
      int coln = wn * 32 + tn * 16 + l;
      int o1 = coln * PITCH + h2 * 8;  // halves K {0..7 | 8..15}, then {16..23 | 24..31}
      bh[tn].q[0] = *(const uint4*)&sBhi[o1];  bh[tn].q[1] = *(const uint4*)&sBhi[o1 + 16];
      bl[tn].q[0] = *(const uint4*)&sBlo[o1];  bl[tn].q[1] = *(const uint4*)&sBlo[o1 + 16];
    }
    #pragma unroll
    for (int tm = 0; tm < 2; ++tm)
      #pragma unroll
      for (int tn = 0; tn < 2; ++tn) {
        acc[tm][tn] = __builtin_amdgcn_wmma_f32_16x16x32_bf16(
            false, ah[tm].v, false, bh[tn].v, (short)0, acc[tm][tn], false, false);
        acc[tm][tn] = __builtin_amdgcn_wmma_f32_16x16x32_bf16(
            false, al[tm].v, false, bh[tn].v, (short)0, acc[tm][tn], false, false);
        acc[tm][tn] = __builtin_amdgcn_wmma_f32_16x16x32_bf16(
            false, ah[tm].v, false, bl[tn].v, (short)0, acc[tm][tn], false, false);
      }
  }

  // Epilogue: bias add, store sal, per-block histogram
  float* salb = sal + (size_t)b * H_ * T_;
  #pragma unroll
  for (int tm = 0; tm < 2; ++tm)
    #pragma unroll
    for (int tn = 0; tn < 2; ++tn)
      #pragma unroll
      for (int g = 0; g < 8; ++g) {
        int row = m0 + wm * 32 + tm * 16 + g + h2 * 8;   // C/D layout: M=g(+8), N=lane&15
        int col = n0 + wn * 32 + tn * 16 + l;
        float v = acc[tm][tn][g] + bsal[row];
        salb[(size_t)row * T_ + col] = v;
        atomicAdd(&sHist[binOf(v)], 1u);
      }
  __syncthreads();
  unsigned* gh = ghist + (size_t)b * NBINS;
  for (int i = tid; i < NBINS; i += 256)
    if (sHist[i]) atomicAdd(&gh[i], sHist[i]);
}

// ---------------- K2: per-batch max + top-k threshold from histogram ----------------
__global__ void scan_hist(const unsigned* __restrict__ hist, float* __restrict__ maxv,
                          int* __restrict__ tbin, int* __restrict__ remain) {
  int b = threadIdx.x;
  if (b >= B_) return;
  const unsigned* h = hist + (size_t)b * NBINS;
  int top = 0;
  for (int i = NBINS - 1; i >= 0; --i) if (h[i]) { top = i; break; }
  maxv[b] = BINLO + (top + 1) * BINW;   // upper edge of top bin: stable softmax shift
  unsigned cum = 0; int tb = 0; int rm = 0;
  for (int i = NBINS - 1; i >= 0; --i) {
    cum += h[i];
    if (cum >= (unsigned)KSP) { tb = i; rm = KSP - (int)(cum - h[i]); break; }
  }
  tbin[b] = tb; remain[b] = rm;
}

// ---------------- K3: per-batch sum of exp(sal - max) ----------------
__global__ __launch_bounds__(256)
void sumexp_kernel(const float* __restrict__ sal, const float* __restrict__ maxv,
                   float* __restrict__ sume) {
  __shared__ float red[256];
  int b = blockIdx.x >> 10;            // 1024 chunks/batch
  int chunk = blockIdx.x & 1023;
  const float* p = sal + ((size_t)b << 23) + ((size_t)chunk * 8192);
  float m = maxv[b];
  float s = 0.0f;
  #pragma unroll
  for (int i = 0; i < 32; ++i) s += __expf(p[i * 256 + threadIdx.x] - m);
  red[threadIdx.x] = s;
  __syncthreads();
  for (int off = 128; off > 0; off >>= 1) {
    if (threadIdx.x < off) red[threadIdx.x] += red[threadIdx.x + off];
    __syncthreads();
  }
  if (threadIdx.x == 0) atomicAdd(&sume[b], red[0]);
}

// ---------------- K4: sparse select + sig dot only at selected positions ----------------
__global__ __launch_bounds__(256)
void finalize_kernel(const float* __restrict__ x, const float* __restrict__ wup,
                     const float* __restrict__ bup, float* __restrict__ out,
                     const float* __restrict__ maxv, const float* __restrict__ sume,
                     const int* __restrict__ tbin, const int* __restrict__ remain,
                     int* __restrict__ tiecnt) {
  size_t base = (size_t)blockIdx.x * 1024;
  #pragma unroll
  for (int j = 0; j < 4; ++j) {
    size_t e = base + (size_t)j * 256 + threadIdx.x;
    int b = (int)(e >> 23);                      // H_*T_ = 2^23
    size_t r = e & ((size_t)(1u << 23) - 1);
    int h = (int)(r >> 13);                      // T_ = 2^13
    int t = (int)(r & (T_ - 1));
    float s = out[e];                            // sal scratch
    int bin = binOf(s);
    float o = 0.0f;
    int tb = tbin[b];
    if (bin >= tb) {
      bool sel = bin > tb;
      if (!sel) sel = atomicAdd(&tiecnt[b], 1) < remain[b];
      if (sel) {
        float val = __expf(s - maxv[b]) / sume[b];
        const float* xr = x + (size_t)b * C_ * T_ + t;
        const float* wr = wup + (size_t)h * C_;
        float d = bup[h];
        #pragma unroll 4
        for (int c = 0; c < C_; ++c) d = fmaf(xr[(size_t)c * T_], wr[c], d);
        o = d * val;
      }
    }
    out[e] = o;
  }
}

// ---------------- host launcher ----------------
extern "C" void kernel_launch(void* const* d_in, const int* in_sizes, int n_in,
                              void* d_out, int out_size, void* d_ws, size_t ws_size,
                              hipStream_t stream) {
  const float* x    = (const float*)d_in[0];
  const float* wup  = (const float*)d_in[1];
  const float* bup  = (const float*)d_in[2];
  const float* wsal = (const float*)d_in[3];
  const float* bsal = (const float*)d_in[4];
  float* out = (float*)d_out;

  char* ws = (char*)d_ws;
  unsigned* hist = (unsigned*)ws;                       // B_*NBINS u32 = 128KB
  float* maxv    = (float*)(ws + B_ * NBINS * 4);
  float* sume    = (float*)(ws + B_ * NBINS * 4 + 32);
  int*   tbin    = (int*)  (ws + B_ * NBINS * 4 + 64);
  int*   remain  = (int*)  (ws + B_ * NBINS * 4 + 96);
  int*   tiecnt  = (int*)  (ws + B_ * NBINS * 4 + 128);

  init_ws<<<dim3((B_ * NBINS + 255) / 256), dim3(256), 0, stream>>>(hist, sume, tiecnt);
  sal_gemm_hist<<<dim3(T_ / 64, H_ / 128, B_), dim3(256), 0, stream>>>(x, wsal, bsal, out, hist);
  scan_hist<<<dim3(1), dim3(32), 0, stream>>>(hist, maxv, tbin, remain);
  sumexp_kernel<<<dim3(B_ * 1024), dim3(256), 0, stream>>>(out, maxv, sume);
  finalize_kernel<<<dim3((B_ * H_ * T_) / 1024), dim3(256), 0, stream>>>(
      x, wup, bup, out, maxv, sume, tbin, remain, tiecnt);
}
